// MODEL_5377299054969
// MI455X (gfx1250) — matrix operand order repaired and checked by reading, not compiled, not run on hardware
//
#include <hip/hip_runtime.h>
#include <hip/hip_bf16.h>
#include <math.h>

// ---------------------------------------------------------------- constants
#define BB 32
#define SS 200
#define NQ 10000
#define ED 256
#define HH 512
#define NF 19
#define ROWS (BB * SS)            // 6400
#define K0P 320                   // 275 padded to multiple of 64 (ping-pong K-step)
#define G4H 2048                  // 4*H
#define PRED_K 531                // H + NF

typedef __attribute__((ext_vector_type(16))) __bf16 v16bf;
typedef __attribute__((ext_vector_type(8)))  __bf16 v8bf;
typedef __attribute__((ext_vector_type(8)))  float  v8f;
typedef __attribute__((vector_size(16)))     int    v4i;

union Frag { v16bf v; v8bf h[2]; };

// CDNA5 async global->LDS path (guarded; falls back to sync staging)
#if defined(__gfx1250__) && __has_builtin(__builtin_amdgcn_global_load_async_to_lds_b128) && __has_builtin(__builtin_amdgcn_s_wait_asynccnt)
#define USE_ASYNC_LDS 1
#else
#define USE_ASYNC_LDS 0
#endif

__device__ __forceinline__ void copy16_g2l(__bf16* ldst, const __bf16* gsrc) {
#if USE_ASYNC_LDS
    __builtin_amdgcn_global_load_async_to_lds_b128(
        (__attribute__((address_space(1))) v4i*)(uintptr_t)gsrc,
        (__attribute__((address_space(3))) v4i*)(unsigned)(uintptr_t)ldst,
        0, 0);
#else
    *(uint4*)ldst = *(const uint4*)gsrc;
#endif
}

__device__ __forceinline__ void lds_stage_fence() {
#if USE_ASYNC_LDS
    __builtin_amdgcn_s_wait_asynccnt(0);
#endif
    __syncthreads();
}

// ISA 7.12.2 bf16 A-frag: two 8-elem chunks at +0 and +16 (kh pre-applied by caller)
__device__ __forceinline__ void load_afrag(Frag& f, const __bf16* base) {
    f.h[0] = *(const v8bf*)(base);
    f.h[1] = *(const v8bf*)(base + 16);
}
// bf16 B-frag: two 8-elem chunks at +0 and +16 (kh*8 pre-applied by caller)
__device__ __forceinline__ void load_bfrag(Frag& f, const __bf16* base) {
    f.h[0] = *(const v8bf*)(base);
    f.h[1] = *(const v8bf*)(base + 16);
}

__device__ __forceinline__ float sigf(float x) { return 1.f / (1.f + __expf(-x)); }

// ---------------------------------------------------------------- prep kernels
__global__ void prep_x0(const int* __restrict__ x_data,
                        const float* __restrict__ embed,
                        const float* __restrict__ Cc,
                        const float* __restrict__ rep,
                        const float* __restrict__ seq,
                        const float* __restrict__ past,
                        __bf16* __restrict__ X0) {
    int idx = blockIdx.x * blockDim.x + threadIdx.x;
    if (idx >= ROWS * K0P) return;
    int row = idx / K0P, k = idx - row * K0P;
    int b = row / SS, t = row - b * SS;
    float v = 0.f;
    if (k < ED) {
        v = embed[(size_t)x_data[row] * ED + k];
    } else if (k < ED + NF) {
        int j = k - ED;
        float cf;
        if (j < 7)        cf = rep [((size_t)b * (SS + 1) + t) * 7 + j];
        else if (j < 13)  cf = seq [((size_t)b * (SS + 1) + t) * 6 + (j - 7)];
        else              cf = past[((size_t)b * (SS + 1) + t) * 6 + (j - 13)];
        v = Cc[(size_t)row * NF + j] * cf;
    }
    X0[idx] = (__bf16)v;
}

__global__ void cvt_weight(const float* __restrict__ W, __bf16* __restrict__ Wb,
                           int N, int Ksrc, int Kdst) {
    int idx = blockIdx.x * blockDim.x + threadIdx.x;
    if (idx >= N * Kdst) return;
    int n = idx / Kdst, k = idx - n * Kdst;
    Wb[idx] = (__bf16)((k < Ksrc) ? W[(size_t)n * Ksrc + k] : 0.f);
}

__global__ void prep_state(const float* __restrict__ init_h,
                           const float* __restrict__ init_c,
                           __bf16* __restrict__ Hinit,
                           float* __restrict__ Cst,
                           float* __restrict__ sums) {
    int idx = blockIdx.x * blockDim.x + threadIdx.x;
    if (idx < 2 * BB * HH) {
        Hinit[idx] = (__bf16)init_h[idx];
        Cst[idx]   = init_c[idx];
    }
    if (idx < 2) sums[idx] = 0.f;
}

// ---------------------------------------------------------------- big batched GEMM
// Out[m,n] = sum_k A[m,k]*W[n,k] + bias1[n] + bias2[n]
// A: [M,K] bf16, W: [N,K] bf16, Out: [M,N] fp32. Block = 8 waves, 32x256 tile.
// A-tile staged in LDS (async on CDNA5); B double-buffered in registers (K-step 64).
__global__ void __launch_bounds__(256)
gemm_bf16_wmma(const __bf16* __restrict__ A,
               const __bf16* __restrict__ W,
               const float* __restrict__ bias1,
               const float* __restrict__ bias2,
               float* __restrict__ Out,
               int N, int K) {
    __shared__ __bf16 smA[32 * 512];
    const int tid  = threadIdx.x;
    const int lane = tid & 31;
    const int w    = tid >> 5;
    const int mb   = blockIdx.y * 32;
    const int n0   = blockIdx.x * 256 + (w & 3) * 64;
    const int mloc = (w >> 2) * 16;
    const int r    = lane & 15;
    const int kh   = lane >> 4;

    // cooperative stage of the 32 x K A-tile (shared by 4 waves per m-tile)
    const int cpr = K >> 3;                      // 16B chunks per row
    for (int c = tid; c < 32 * cpr; c += 256) {
        int rr = c / cpr, cc = c - rr * cpr;
        copy16_g2l(&smA[rr * K + cc * 8], &A[(size_t)(mb + rr) * K + cc * 8]);
    }
    lds_stage_fence();

    const v8f vz = {0.f, 0.f, 0.f, 0.f, 0.f, 0.f, 0.f, 0.f};
    v8f acc[4] = {vz, vz, vz, vz};
    Frag bA[4], bB[4];
    const __bf16* wbase = W + (size_t)(n0 + r) * K + kh * 8;
    const __bf16* abase = &smA[(mloc + r) * K + kh * 8];

#pragma unroll
    for (int s = 0; s < 4; ++s) load_bfrag(bA[s], wbase + (size_t)s * 16 * K);

    for (int k0 = 0; k0 < K; k0 += 64) {
#pragma unroll
        for (int s = 0; s < 4; ++s) load_bfrag(bB[s], wbase + (size_t)s * 16 * K + k0 + 32);
        Frag a0; load_afrag(a0, abase + k0);
#pragma unroll
        for (int s = 0; s < 4; ++s)
            acc[s] = __builtin_amdgcn_wmma_f32_16x16x32_bf16(
                false, a0.v, false, bA[s].v, (short)0, acc[s], false, false);
        if (k0 + 64 < K) {
#pragma unroll
            for (int s = 0; s < 4; ++s) load_bfrag(bA[s], wbase + (size_t)s * 16 * K + k0 + 64);
        }
        Frag a1; load_afrag(a1, abase + k0 + 32);
#pragma unroll
        for (int s = 0; s < 4; ++s)
            acc[s] = __builtin_amdgcn_wmma_f32_16x16x32_bf16(
                false, a1.v, false, bB[s].v, (short)0, acc[s], false, false);
    }
#pragma unroll
    for (int s = 0; s < 4; ++s) {
        int n = n0 + s * 16 + (lane & 15);
        float bs = bias1[n] + bias2[n];
#pragma unroll
        for (int rr = 0; rr < 8; ++rr) {
            int m = mb + mloc + rr + 8 * kh;
            Out[(size_t)m * N + n] = acc[s][rr] + bs;
        }
    }
}

// ---------------------------------------------------------------- fused LSTM step
// One timestep: R = Hprev @ Whh^T for all 4 gates, g = G_t + R, cell update.
// Grid: 8 blocks x 256 threads; block stages its 16 x 512 h_prev tile in LDS
// (shared by all 8 waves); per-gate B fragments double-buffered in registers.
__global__ void __launch_bounds__(256)
lstm_step(const __bf16* __restrict__ Hprev, size_t hstride,
          const __bf16* __restrict__ Whh,   // [2048, 512]
          const float* __restrict__ G,      // [ROWS, 2048], row b*SS+t
          float* __restrict__ Cst,          // [BB, HH] this layer
          __bf16* __restrict__ Hout,        // row (b*SS+t)
          int t) {
    __shared__ __bf16 smH[16 * HH];
    const int tid  = threadIdx.x;
    const int lane = tid & 31;
    const int w    = tid >> 5;
    const int m0   = (blockIdx.x >> 2) * 16;                 // block-level batch tile
    const int j0   = (((blockIdx.x & 3) << 3) + w) * 16;     // 16 hidden units per wave
    const int r    = lane & 15;
    const int kh   = lane >> 4;

    // stage h_prev tile (16 rows x 512) — read once instead of 8x per block
    for (int c = tid; c < 16 * (HH >> 3); c += 256) {
        int rr = c >> 6, cc = c & 63;
        copy16_g2l(&smH[rr * HH + cc * 8], &Hprev[(size_t)(m0 + rr) * hstride + cc * 8]);
    }
    lds_stage_fence();

    const v8f vz = {0.f, 0.f, 0.f, 0.f, 0.f, 0.f, 0.f, 0.f};
    v8f acc[4] = {vz, vz, vz, vz};                           // i, f, g, o
    Frag bA[4], bB[4];
    const __bf16* wbase = Whh + (size_t)(j0 + r) * HH + kh * 8;
    const __bf16* abase = &smH[r * HH + kh * 8];

#pragma unroll
    for (int g = 0; g < 4; ++g) load_bfrag(bA[g], wbase + (size_t)g * HH * HH);

    for (int k0 = 0; k0 < HH; k0 += 64) {
#pragma unroll
        for (int g = 0; g < 4; ++g) load_bfrag(bB[g], wbase + (size_t)g * HH * HH + k0 + 32);
        Frag a0; load_afrag(a0, abase + k0);
#pragma unroll
        for (int g = 0; g < 4; ++g)
            acc[g] = __builtin_amdgcn_wmma_f32_16x16x32_bf16(
                false, a0.v, false, bA[g].v, (short)0, acc[g], false, false);
        if (k0 + 64 < HH) {
#pragma unroll
            for (int g = 0; g < 4; ++g) load_bfrag(bA[g], wbase + (size_t)g * HH * HH + k0 + 64);
        }
        Frag a1; load_afrag(a1, abase + k0 + 32);
#pragma unroll
        for (int g = 0; g < 4; ++g)
            acc[g] = __builtin_amdgcn_wmma_f32_16x16x32_bf16(
                false, a1.v, false, bB[g].v, (short)0, acc[g], false, false);
    }

    const int j = j0 + (lane & 15);
#pragma unroll
    for (int rr = 0; rr < 8; ++rr) {
        int b = m0 + rr + 8 * kh;
        const float* grow = G + ((size_t)b * SS + t) * G4H;
        float gi = acc[0][rr] + grow[j];
        float gf = acc[1][rr] + grow[HH + j];
        float gg = acc[2][rr] + grow[2 * HH + j];
        float go = acc[3][rr] + grow[3 * HH + j];
        float cold = Cst[b * HH + j];
        float cn = sigf(gf) * cold + sigf(gi) * tanhf(gg);
        float hn = sigf(go) * tanhf(cn);
        Cst[b * HH + j] = cn;
        Hout[((size_t)b * SS + t) * HH + j] = (__bf16)hn;
    }
}

// ---------------------------------------------------------------- gathered prediction + BCE
__global__ void pred_kernel(const __bf16* __restrict__ H1,
                            const int* __restrict__ q_t,
                            const float* __restrict__ target,
                            const float* __restrict__ Wp,   // [NQ+1, 531]
                            const float* __restrict__ bp,
                            const float* __restrict__ Cc,
                            const float* __restrict__ rep,
                            const float* __restrict__ seq,
                            const float* __restrict__ past,
                            float* __restrict__ sums,
                            float* __restrict__ out) {
    int row  = blockIdx.x * 8 + (threadIdx.x >> 5);
    if (row >= ROWS) return;
    int lane = threadIdx.x & 31;
    int q    = q_t[row];
    const float* wrow = Wp + (size_t)q * PRED_K;

    float p = 0.f;
    for (int k = lane; k < HH; k += 32)
        p += (float)H1[(size_t)row * HH + k] * wrow[k];
    if (lane < NF) {
        int b = row / SS, t = row - b * SS, j = lane;
        float cf;
        if (j < 7)       cf = rep [((size_t)b * (SS + 1) + t + 1) * 7 + j];
        else if (j < 13) cf = seq [((size_t)b * (SS + 1) + t + 1) * 6 + (j - 7)];
        else             cf = past[((size_t)b * (SS + 1) + t + 1) * 6 + (j - 13)];
        p += Cc[(size_t)row * NF + j] * cf * wrow[HH + j];
    }
#pragma unroll
    for (int off = 16; off > 0; off >>= 1) p += __shfl_down(p, off, 32);

    if (lane == 0) {
        float x   = p + bp[q];
        float m   = (q > 0) ? 1.f : 0.f;
        float tgt = target[row];
        float bce = fmaxf(x, 0.f) - x * tgt + log1pf(__expf(-fabsf(x)));
        atomicAdd(&sums[0], bce * m);
        atomicAdd(&sums[1], m);
        out[1 + row]        = m * sigf(x);
        out[1 + ROWS + row] = tgt * m;
    }
}

__global__ void finalize(const float* __restrict__ sums, float* __restrict__ out) {
    out[0] = sums[0] / sums[1];
}

// ---------------------------------------------------------------- host launch
extern "C" void kernel_launch(void* const* d_in, const int* in_sizes, int n_in,
                              void* d_out, int out_size, void* d_ws, size_t ws_size,
                              hipStream_t stream) {
    const int*   x_data = (const int*)d_in[0];
    const int*   q_t    = (const int*)d_in[1];
    const float* target = (const float*)d_in[2];
    const float* rep    = (const float*)d_in[3];
    const float* past   = (const float*)d_in[4];
    const float* seq    = (const float*)d_in[5];
    const float* embed  = (const float*)d_in[6];
    const float* W_ih0  = (const float*)d_in[7];
    const float* W_hh0  = (const float*)d_in[8];
    const float* b_ih0  = (const float*)d_in[9];
    const float* b_hh0  = (const float*)d_in[10];
    const float* W_ih1  = (const float*)d_in[11];
    const float* W_hh1  = (const float*)d_in[12];
    const float* b_ih1  = (const float*)d_in[13];
    const float* b_hh1  = (const float*)d_in[14];
    const float* W_pred = (const float*)d_in[15];
    const float* b_pred = (const float*)d_in[16];
    const float* Cc     = (const float*)d_in[17];
    const float* init_h = (const float*)d_in[18];
    const float* init_c = (const float*)d_in[19];
    float* out = (float*)d_out;

    // workspace layout (256B aligned regions)
    char* ws = (char*)d_ws;
    size_t off = 0;
    auto alloc = [&](size_t bytes) { char* p = ws + off; off = (off + bytes + 255) & ~(size_t)255; return p; };
    __bf16* X0     = (__bf16*)alloc((size_t)ROWS * K0P * 2);
    __bf16* Wih0b  = (__bf16*)alloc((size_t)G4H * K0P * 2);
    __bf16* Whh0b  = (__bf16*)alloc((size_t)G4H * HH * 2);
    __bf16* Wih1b  = (__bf16*)alloc((size_t)G4H * HH * 2);
    __bf16* Whh1b  = (__bf16*)alloc((size_t)G4H * HH * 2);
    float*  G      = (float*)alloc((size_t)ROWS * G4H * 4);   // reused for both layers
    __bf16* X1     = (__bf16*)alloc((size_t)ROWS * HH * 2);   // layer-0 hidden (bf16)
    __bf16* H1     = (__bf16*)alloc((size_t)ROWS * HH * 2);   // layer-1 hidden (bf16)
    __bf16* Hinit  = (__bf16*)alloc((size_t)2 * BB * HH * 2);
    float*  Cst    = (float*)alloc((size_t)2 * BB * HH * 4);
    float*  sums   = (float*)alloc(2 * 4);

    // --- prep
    prep_x0<<<(ROWS * K0P + 255) / 256, 256, 0, stream>>>(x_data, embed, Cc, rep, seq, past, X0);
    cvt_weight<<<(G4H * K0P + 255) / 256, 256, 0, stream>>>(W_ih0, Wih0b, G4H, ED + NF, K0P);
    cvt_weight<<<(G4H * HH + 255) / 256, 256, 0, stream>>>(W_hh0, Whh0b, G4H, HH, HH);
    cvt_weight<<<(G4H * HH + 255) / 256, 256, 0, stream>>>(W_ih1, Wih1b, G4H, HH, HH);
    cvt_weight<<<(G4H * HH + 255) / 256, 256, 0, stream>>>(W_hh1, Whh1b, G4H, HH, HH);
    prep_state<<<(2 * BB * HH + 255) / 256, 256, 0, stream>>>(init_h, init_c, Hinit, Cst, sums);

    dim3 gg(G4H / 256, ROWS / 32);

    // --- layer 0: batched input GEMM, then 200 fused recurrent steps
    gemm_bf16_wmma<<<gg, 256, 0, stream>>>(X0, Wih0b, b_ih0, b_hh0, G, G4H, K0P);
    for (int t = 0; t < SS; ++t) {
        const __bf16* hp = (t == 0) ? Hinit : X1 + (size_t)(t - 1) * HH;
        size_t rs = (t == 0) ? (size_t)HH : (size_t)SS * HH;
        lstm_step<<<8, 256, 0, stream>>>(hp, rs, Whh0b, G, Cst, X1, t);
    }

    // --- layer 1
    gemm_bf16_wmma<<<gg, 256, 0, stream>>>(X1, Wih1b, b_ih1, b_hh1, G, G4H, HH);
    for (int t = 0; t < SS; ++t) {
        const __bf16* hp = (t == 0) ? Hinit + (size_t)BB * HH : H1 + (size_t)(t - 1) * HH;
        size_t rs = (t == 0) ? (size_t)HH : (size_t)SS * HH;
        lstm_step<<<8, 256, 0, stream>>>(hp, rs, Whh1b, G, Cst + (size_t)BB * HH, H1, t);
    }

    // --- gathered prediction + masked BCE
    pred_kernel<<<ROWS / 8, 256, 0, stream>>>(H1, q_t, target, W_pred, b_pred,
                                              Cc, rep, seq, past, sums, out);
    finalize<<<1, 1, 0, stream>>>(sums, out);
    (void)in_sizes; (void)n_in; (void)out_size; (void)ws_size;
}
